// MiniGPT2_10599979286826
// MI455X (gfx1250) — compile-verified, matrix operand order checked by reading, not run on hardware
//
#include <hip/hip_runtime.h>
#include <cmath>
#include <cstdint>

#define DEV __device__ __forceinline__

typedef __attribute__((ext_vector_type(16))) __bf16 v16bf;
typedef __attribute__((ext_vector_type(8)))  float  v8f;

// ---- model dims (fixed) ----
constexpr int B_  = 2;
constexpr int T_  = 1024;
constexpr int D_  = 1024;
constexpr int H_  = 16;
constexpr int HD_ = 64;
constexpr int L_  = 8;
constexpr int FF_ = 4096;
constexpr int V_  = 50257;
constexpr int BT_ = B_ * T_;
constexpr float SCALE_ = 0.125f; // 1/sqrt(64)

// ---------- bf16 helpers (storage = unsigned short) ----------
DEV unsigned short f2bf(float f) {
  unsigned int x = __builtin_bit_cast(unsigned int, f);
  x += 0x7FFFu + ((x >> 16) & 1u);               // round-to-nearest-even
  return (unsigned short)(x >> 16);
}
DEV __bf16 u2bf(unsigned short u) { return __builtin_bit_cast(__bf16, u); }

// ---------- WMMA fragment loaders (wave32, 16x16x32 bf16) ----------
// A operand 16x32: element e of lane maps to K = (e&7) + ((e>>3)<<4) + (lane>>4)*8,
// row = lane&15 (two contiguous 8-bf16 groups = two 16B loads per lane).
// B operand for C = A @ W^T with W row-major [N x K] has the identical layout.
DEV v16bf load_frag_rowK(const unsigned short* base, int ld, int row0, int k0,
                         int rowmax) {
  const int lane = threadIdx.x & 31;
  int row = row0 + (lane & 15);
  if (row > rowmax) row = rowmax;                 // clamp (ragged N for logits)
  const int hi = lane >> 4;
  const unsigned short* p = base + (size_t)row * ld + k0 + hi * 8;
  v16bf v;
#pragma unroll
  for (int e = 0; e < 8; ++e) v[e] = u2bf(p[e]);
#pragma unroll
  for (int e = 0; e < 8; ++e) v[8 + e] = u2bf(p[16 + e]);
  return v;
}

// Same fragment sourced from an LDS tile laid out [16*mt rows][32 cols].
DEV v16bf load_frag_lds(const unsigned short* tile, int row0) {
  const int lane = threadIdx.x & 31;
  const unsigned short* p = tile + (row0 + (lane & 15)) * 32 + (lane >> 4) * 8;
  v16bf v;
#pragma unroll
  for (int e = 0; e < 8; ++e) v[e] = u2bf(p[e]);
#pragma unroll
  for (int e = 0; e < 8; ++e) v[8 + e] = u2bf(p[16 + e]);
  return v;
}

// B operand when B is stored row-major [K x N] (NOT transposed): element
// (k, n) lives at base[k*ld + n]; lane carries column n, elems carry K.
DEV v16bf load_frag_colN(const unsigned short* base, int ld, int k0, int col0) {
  const int lane = threadIdx.x & 31;
  const int col = col0 + (lane & 15);
  const int hi = lane >> 4;
  v16bf v;
#pragma unroll
  for (int e = 0; e < 16; ++e) {
    const int k = (e & 7) + ((e >> 3) << 4) + hi * 8;
    v[e] = u2bf(base[(size_t)(k0 + k) * ld + col]);
  }
  return v;
}

DEV v8f wmma_bf16(v16bf a, v16bf b, v8f c) {
  return __builtin_amdgcn_wmma_f32_16x16x32_bf16(false, a, false, b,
                                                 (short)0, c, false, false);
}

// ---------- CDNA5 async global->LDS copy (ASYNCcnt path) ----------
// Per CDNA5 ISA: GLOBAL_LOAD_ASYNC_TO_LDS_B128, GV mode.
// LDS byte address from VDST VGPR; flat LDS addresses map via addr[31:0].
DEV void async_copy_b128(unsigned lds_byte_addr, const void* gptr) {
  asm volatile("global_load_async_to_lds_b128 %0, %1, off"
               :: "v"(lds_byte_addr),
                  "v"((unsigned long long)(uintptr_t)gptr)
               : "memory");
}
DEV void wait_async_le1() { asm volatile("s_wait_asynccnt 0x1" ::: "memory"); }
DEV void wait_async_le0() { asm volatile("s_wait_asynccnt 0x0" ::: "memory"); }

// ---------- elementwise kernels ----------
__global__ __launch_bounds__(256) void k_f32_to_bf16(const float* __restrict__ s,
                                                     unsigned short* __restrict__ d,
                                                     size_t n4) {
  size_t i = (size_t)blockIdx.x * 256 + threadIdx.x;
  if (i >= n4) return;
  float4 v = reinterpret_cast<const float4*>(s)[i];
  size_t o = i * 4;
  d[o + 0] = f2bf(v.x); d[o + 1] = f2bf(v.y);
  d[o + 2] = f2bf(v.z); d[o + 3] = f2bf(v.w);
}

__global__ __launch_bounds__(256) void k_embed(const int* __restrict__ idx,
                                               const float* __restrict__ tok,
                                               const float* __restrict__ pos,
                                               float* __restrict__ x) {
  const int row = blockIdx.x;            // row in [0, B*T)
  const int t = row % T_;
  const int tokid = idx[row];
  const int d = threadIdx.x * 4;         // 256 threads * 4 = D
  const float4 te = *reinterpret_cast<const float4*>(tok + (size_t)tokid * D_ + d);
  const float4 pe = *reinterpret_cast<const float4*>(pos + (size_t)t * D_ + d);
  float4 o; o.x = te.x + pe.x; o.y = te.y + pe.y; o.z = te.z + pe.z; o.w = te.w + pe.w;
  *reinterpret_cast<float4*>(x + (size_t)row * D_ + d) = o;
}

__global__ __launch_bounds__(256) void k_layernorm_bf(const float* __restrict__ x,
                                                      const float* __restrict__ w,
                                                      const float* __restrict__ b,
                                                      unsigned short* __restrict__ out) {
  __shared__ float red[256];
  const int row = blockIdx.x;
  const int tid = threadIdx.x;
  const float* xr = x + (size_t)row * D_;
  const float4 v = reinterpret_cast<const float4*>(xr)[tid];

  float s = v.x + v.y + v.z + v.w;
  red[tid] = s; __syncthreads();
  for (int off = 128; off > 0; off >>= 1) {
    if (tid < off) red[tid] += red[tid + off];
    __syncthreads();
  }
  const float mu = red[0] * (1.0f / D_);
  __syncthreads();

  float4 dv; dv.x = v.x - mu; dv.y = v.y - mu; dv.z = v.z - mu; dv.w = v.w - mu;
  red[tid] = dv.x * dv.x + dv.y * dv.y + dv.z * dv.z + dv.w * dv.w;
  __syncthreads();
  for (int off = 128; off > 0; off >>= 1) {
    if (tid < off) red[tid] += red[tid + off];
    __syncthreads();
  }
  const float rs = rsqrtf(red[0] * (1.0f / D_) + 1e-5f);

  const int d = tid * 4;
  const float4 wv = *reinterpret_cast<const float4*>(w + d);
  const float4 bv = *reinterpret_cast<const float4*>(b + d);
  unsigned short* o = out + (size_t)row * D_ + d;
  o[0] = f2bf(dv.x * rs * wv.x + bv.x);
  o[1] = f2bf(dv.y * rs * wv.y + bv.y);
  o[2] = f2bf(dv.z * rs * wv.z + bv.z);
  o[3] = f2bf(dv.w * rs * wv.w + bv.w);
}

// ---------- GEMM: C[M,N] = A[M,K] @ W[N,K]^T (+epilogue) ----------
// Block = 256 threads = 8 waves. Block tile: 64 (M) x 512 (N); wave w owns the
// 64x64 strip at n0 = blockX*512 + w*64 with a 4x4 grid of 16x16 WMMA tiles
// (16 f32 accumulators). The 64x32 A slab per K-step is async-copied into a
// double-buffered LDS tile once per block and shared by all 8 waves (8x A
// traffic reduction); B streams from global (L2-resident weight slabs).
template <int EPI>
__global__ __launch_bounds__(256) void gemm_bf16(const unsigned short* __restrict__ A,
                                                 const unsigned short* __restrict__ W,
                                                 const float* __restrict__ bias,
                                                 const float* __restrict__ resid,
                                                 float* __restrict__ Cf,
                                                 unsigned short* __restrict__ Cbf,
                                                 int M, int N, int K) {
  __shared__ unsigned short As[2][64 * 32];      // 2 x 4KB double buffer

  const int tid  = threadIdx.x;
  const int wave = tid >> 5;
  const int lane = tid & 31;
  const int m0 = blockIdx.y * 64;
  const int n0 = blockIdx.x * 512 + wave * 64;
  const bool active = (n0 < N);                  // wave-uniform

  // async-copy assignment: thread tid moves 16B; 256*16B = 64x32 bf16 slab
  const int crow = tid >> 2;                     // 0..63
  const int ccol = (tid & 3) * 8;                // 0,8,16,24
  const unsigned ldsb = (unsigned)(uintptr_t)&As[0][0];

  const int nk = K / 32;
  // prologue: stage k-step 0 into buffer 0
  async_copy_b128(ldsb + (unsigned)(crow * 32 + ccol) * 2,
                  A + (size_t)(m0 + crow) * K + ccol);

  v8f acc[4][4] = {};
  for (int ki = 0; ki < nk; ++ki) {
    const int k0 = ki * 32;
    const bool more = (ki + 1 < nk);
    if (more)  // stage next slab into the other buffer (read-done last iter)
      async_copy_b128(ldsb + (unsigned)(((ki + 1) & 1) * (64 * 32) + crow * 32 + ccol) * 2,
                      A + (size_t)(m0 + crow) * K + k0 + 32 + ccol);
    if (more) wait_async_le1(); else wait_async_le0();   // current slab landed
    __syncthreads();

    if (active) {
      const unsigned short* Abuf = &As[ki & 1][0];
      v16bf a[4];
#pragma unroll
      for (int mi = 0; mi < 4; ++mi) a[mi] = load_frag_lds(Abuf, mi * 16);
#pragma unroll
      for (int nj = 0; nj < 4; ++nj) {
        if (n0 + nj * 16 < N) {                  // wave-uniform
          if (more)
            __builtin_prefetch(W + (size_t)(n0 + nj * 16 + (lane & 15)) * K + k0 + 32, 0, 0);
          const v16bf b = load_frag_rowK(W, K, n0 + nj * 16, k0, N - 1);
#pragma unroll
          for (int mi = 0; mi < 4; ++mi)
            acc[mi][nj] = wmma_bf16(a[mi], b, acc[mi][nj]);
        }
      }
    }
    __syncthreads();                             // protect buffer for next copy
  }

  if (!active) return;
  const int nn = lane & 15;
  const int mb = (lane >> 4) * 8;
#pragma unroll
  for (int mi = 0; mi < 4; ++mi) {
#pragma unroll
    for (int nj = 0; nj < 4; ++nj) {
#pragma unroll
      for (int r = 0; r < 8; ++r) {
        const int m = m0 + mi * 16 + mb + r;
        const int n = n0 + nj * 16 + nn;
        if (m < M && n < N) {
          float v = acc[mi][nj][r];
          if (bias) v += bias[n];
          if (EPI == 1) v += resid[(size_t)m * N + n];
          if (EPI == 2) v = 0.5f * v * (1.0f + erff(v * 0.70710678118f));
          if (Cf)  Cf[(size_t)m * N + n] = v;
          if (Cbf) Cbf[(size_t)m * N + n] = f2bf(v);
        }
      }
    }
  }
}

// ---------- attention: S = scale * Q @ K^T (per b,h; skip above-diagonal) ----
__global__ __launch_bounds__(256) void k_attn_scores(const unsigned short* __restrict__ qkv,
                                                     float* __restrict__ scores) {
  const int wave = threadIdx.x >> 5;
  const int lane = threadIdx.x & 31;
  const int j = blockIdx.x * 8 + wave;   // k-tile
  const int i = blockIdx.y;              // q-tile
  if (j > i) return;                     // causal: never read by softmax
  const int bh = blockIdx.z;
  const int b = bh / H_, h = bh % H_;

  const unsigned short* Qb = qkv + (size_t)b * T_ * 3 * D_ + 0 * D_ + h * HD_;
  const unsigned short* Kb = qkv + (size_t)b * T_ * 3 * D_ + 1 * D_ + h * HD_;

  v8f acc = {};
#pragma unroll
  for (int k0 = 0; k0 < HD_; k0 += 32) {
    const v16bf a  = load_frag_rowK(Qb, 3 * D_, i * 16, k0, T_ - 1);
    const v16bf bm = load_frag_rowK(Kb, 3 * D_, j * 16, k0, T_ - 1);
    acc = wmma_bf16(a, bm, acc);
  }

  const int nn = lane & 15;
  const int mb = (lane >> 4) * 8;
  float* srow = scores + (size_t)bh * T_ * T_;
#pragma unroll
  for (int r = 0; r < 8; ++r) {
    const int q = i * 16 + mb + r;
    const int k = j * 16 + nn;
    srow[(size_t)q * T_ + k] = acc[r] * SCALE_;
  }
}

// ---------- causal softmax over valid prefix; probs -> bf16 (zero beyond q) --
__global__ __launch_bounds__(256) void k_softmax(const float* __restrict__ scores,
                                                 unsigned short* __restrict__ pbf) {
  __shared__ float red[256];
  const int row = blockIdx.x;            // (b*H + h)*T + q
  const int tid = threadIdx.x;
  const int q = row % T_;
  const int valid = q + 1;
  const float* s = scores + (size_t)row * T_;
  unsigned short* p = pbf + (size_t)row * T_;

  float mx = -3.4e38f;
  for (int i = tid; i < valid; i += 256) mx = fmaxf(mx, s[i]);
  red[tid] = mx; __syncthreads();
  for (int off = 128; off > 0; off >>= 1) {
    if (tid < off) red[tid] = fmaxf(red[tid], red[tid + off]);
    __syncthreads();
  }
  mx = red[0]; __syncthreads();

  float sum = 0.0f;
  for (int i = tid; i < valid; i += 256) sum += __expf(s[i] - mx);
  red[tid] = sum; __syncthreads();
  for (int off = 128; off > 0; off >>= 1) {
    if (tid < off) red[tid] += red[tid + off];
    __syncthreads();
  }
  const float inv = 1.0f / red[0];

  for (int i = tid; i < T_; i += 256)
    p[i] = (i < valid) ? f2bf(__expf(s[i] - mx) * inv) : (unsigned short)0;
}

// ---------- Y = P @ V (per b,h). B operand is V, K-strided. ------------------
__global__ __launch_bounds__(256) void k_attn_pv(const unsigned short* __restrict__ pbf,
                                                 const unsigned short* __restrict__ qkv,
                                                 unsigned short* __restrict__ ybf) {
  const int wave = threadIdx.x >> 5;
  const int lane = threadIdx.x & 31;
  const int i = blockIdx.x * 8 + wave;   // q-tile
  if (i >= T_ / 16) return;
  const int bh = blockIdx.z;
  const int b = bh / H_, h = bh % H_;

  const unsigned short* Pb = pbf + (size_t)bh * T_ * T_;
  const unsigned short* Vb = qkv + (size_t)b * T_ * 3 * D_ + 2 * D_ + h * HD_;

  v8f acc[4] = {};
  const int kend = i * 16 + 16;          // probs are zero for k > q
  for (int k0 = 0; k0 < kend; k0 += 32) {
    const v16bf a = load_frag_rowK(Pb, T_, i * 16, k0, T_ - 1);
#pragma unroll
    for (int j = 0; j < 4; ++j) {        // HD = 64 -> 4 n-tiles
      const v16bf bm = load_frag_colN(Vb, 3 * D_, k0, j * 16);
      acc[j] = wmma_bf16(a, bm, acc[j]);
    }
  }

  const int nn = lane & 15;
  const int mb = (lane >> 4) * 8;
#pragma unroll
  for (int j = 0; j < 4; ++j) {
#pragma unroll
    for (int r = 0; r < 8; ++r) {
      const int q = i * 16 + mb + r;
      const int hd = j * 16 + nn;
      ybf[((size_t)b * T_ + q) * D_ + h * HD_ + hd] = f2bf(acc[j][r]);
    }
  }
}

// =================== host-side orchestration ===================
extern "C" void kernel_launch(void* const* d_in, const int* in_sizes, int n_in,
                              void* d_out, int out_size, void* d_ws, size_t ws_size,
                              hipStream_t stream) {
  const int*   idx     = (const int*)  d_in[0];
  const float* tok_emb = (const float*)d_in[1];
  const float* pos_emb = (const float*)d_in[2];
  const float* ln1_w   = (const float*)d_in[3];
  const float* ln1_b   = (const float*)d_in[4];
  const float* qkv_w   = (const float*)d_in[5];
  const float* qkv_b   = (const float*)d_in[6];
  const float* proj_w  = (const float*)d_in[7];
  const float* proj_b  = (const float*)d_in[8];
  const float* ln2_w   = (const float*)d_in[9];
  const float* ln2_b   = (const float*)d_in[10];
  const float* fc1_w   = (const float*)d_in[11];
  const float* fc1_b   = (const float*)d_in[12];
  const float* fc2_w   = (const float*)d_in[13];
  const float* fc2_b   = (const float*)d_in[14];
  const float* lnf_w   = (const float*)d_in[15];
  const float* lnf_b   = (const float*)d_in[16];
  float* logits = (float*)d_out;

  // workspace carve (bump allocator, 256B aligned)
  char* p = (char*)d_ws;
  auto alloc = [&](size_t bytes) -> void* {
    void* r = (void*)p;
    p += (bytes + 255) & ~(size_t)255;
    return r;
  };
  float*          x      = (float*)         alloc((size_t)BT_ * D_ * 4);
  unsigned short* h_bf   = (unsigned short*)alloc((size_t)BT_ * D_ * 2);
  unsigned short* qkv_bf = (unsigned short*)alloc((size_t)BT_ * 3 * D_ * 2);
  float*          scores = (float*)         alloc((size_t)B_ * H_ * T_ * T_ * 4);
  unsigned short* pbf    = (unsigned short*)alloc((size_t)B_ * H_ * T_ * T_ * 2);
  unsigned short* ybf    = (unsigned short*)alloc((size_t)BT_ * D_ * 2);
  unsigned short* ff_bf  = (unsigned short*)alloc((size_t)BT_ * FF_ * 2);
  unsigned short* wq_bf  = (unsigned short*)alloc((size_t)L_ * 3 * D_ * D_ * 2);
  unsigned short* wp_bf  = (unsigned short*)alloc((size_t)L_ * D_ * D_ * 2);
  unsigned short* w1_bf  = (unsigned short*)alloc((size_t)L_ * FF_ * D_ * 2);
  unsigned short* w2_bf  = (unsigned short*)alloc((size_t)L_ * D_ * FF_ * 2);
  unsigned short* wt_bf  = (unsigned short*)alloc((size_t)V_ * D_ * 2);
  (void)ws_size; (void)in_sizes; (void)n_in; (void)out_size;

  auto convert = [&](const float* src, unsigned short* dst, size_t n) {
    size_t n4 = n / 4;
    k_f32_to_bf16<<<dim3((unsigned)((n4 + 255) / 256)), 256, 0, stream>>>(src, dst, n4);
  };
  convert(qkv_w,   wq_bf, (size_t)L_ * 3 * D_ * D_);
  convert(proj_w,  wp_bf, (size_t)L_ * D_ * D_);
  convert(fc1_w,   w1_bf, (size_t)L_ * FF_ * D_);
  convert(fc2_w,   w2_bf, (size_t)L_ * D_ * FF_);
  convert(tok_emb, wt_bf, (size_t)V_ * D_);

  k_embed<<<dim3(BT_), 256, 0, stream>>>(idx, tok_emb, pos_emb, x);

  auto gemm = [&](int epi, const unsigned short* A, const unsigned short* W,
                  const float* bias, const float* resid, float* Cf,
                  unsigned short* Cbf, int M, int N, int K) {
    dim3 grid((unsigned)((N + 511) / 512), (unsigned)(M / 64));
    if (epi == 0)      gemm_bf16<0><<<grid, 256, 0, stream>>>(A, W, bias, resid, Cf, Cbf, M, N, K);
    else if (epi == 1) gemm_bf16<1><<<grid, 256, 0, stream>>>(A, W, bias, resid, Cf, Cbf, M, N, K);
    else               gemm_bf16<2><<<grid, 256, 0, stream>>>(A, W, bias, resid, Cf, Cbf, M, N, K);
  };

  for (int l = 0; l < L_; ++l) {
    // ln1 -> h_bf
    k_layernorm_bf<<<dim3(BT_), 256, 0, stream>>>(x, ln1_w + (size_t)l * D_,
                                                  ln1_b + (size_t)l * D_, h_bf);
    // qkv = h @ Wq^T + bq  (bf16 out only)
    gemm(0, h_bf, wq_bf + (size_t)l * 3 * D_ * D_, qkv_b + (size_t)l * 3 * D_,
         nullptr, nullptr, qkv_bf, BT_, 3 * D_, D_);
    // S = scale * Q K^T  (lower-triangular tiles only)
    k_attn_scores<<<dim3(T_ / 16 / 8, T_ / 16, B_ * H_), 256, 0, stream>>>(qkv_bf, scores);
    // causal softmax -> bf16 probs (zero-padded beyond q)
    k_softmax<<<dim3(B_ * H_ * T_), 256, 0, stream>>>(scores, pbf);
    // Y = P V
    k_attn_pv<<<dim3(T_ / 16 / 8, 1, B_ * H_), 256, 0, stream>>>(pbf, qkv_bf, ybf);
    // x = x + Y @ Wp^T + bp   (in-place residual)
    gemm(1, ybf, wp_bf + (size_t)l * D_ * D_, proj_b + (size_t)l * D_,
         x, x, nullptr, BT_, D_, D_);
    // ln2 -> h_bf
    k_layernorm_bf<<<dim3(BT_), 256, 0, stream>>>(x, ln2_w + (size_t)l * D_,
                                                  ln2_b + (size_t)l * D_, h_bf);
    // ff = gelu(h @ W1^T + b1) -> bf16
    gemm(2, h_bf, w1_bf + (size_t)l * FF_ * D_, fc1_b + (size_t)l * FF_,
         nullptr, nullptr, ff_bf, BT_, FF_, D_);
    // x = x + ff @ W2^T + b2
    gemm(1, ff_bf, w2_bf + (size_t)l * D_ * FF_, fc2_b + (size_t)l * D_,
         x, x, nullptr, BT_, D_, FF_);
  }

  // final ln -> h_bf ; logits = h @ tok_emb^T (no bias)
  k_layernorm_bf<<<dim3(BT_), 256, 0, stream>>>(x, lnf_w, lnf_b, h_bf);
  gemm(0, h_bf, wt_bf, nullptr, nullptr, logits, nullptr, BT_, V_, D_);
}
